// MultiHeadRelativeAttention_6562710028732
// MI455X (gfx1250) — compile-verified, matrix operand order checked by reading, not run on hardware
//
#include <hip/hip_runtime.h>
#include <hip/hip_bf16.h>
#include <cstdint>

// Problem constants (from setup_inputs): B=2, S=2048, D=512, H=8, hd=64,
// MAX_REL=512, NUM_RPRS=1025
#define BN  2
#define SN  2048
#define DN  512
#define HN  8
#define HDN 64
#define NRN 1025
#define MRN 512

typedef __attribute__((ext_vector_type(16))) __bf16 v16bf;
typedef __attribute__((ext_vector_type(8)))  float  v8f;
typedef __attribute__((ext_vector_type(4)))  int    v4i;

// ---- CDNA5 async global->LDS copy (ASYNCcnt path), feature-guarded --------
#if defined(__has_builtin)
#  if __has_builtin(__builtin_amdgcn_global_load_async_to_lds_b128)
#    define USE_ASYNC_LDS 1
#  endif
#endif

// Builtin signature (from compiler diagnostic): param1 = v4i in AS1 (global,
// prints as "__device__"), param2 = v4i in AS3 (LDS), then imm offset, imm cpol.
#define AS1P(p) ((__attribute__((address_space(1))) v4i*)(uintptr_t)(p))
#define AS3P(p) ((__attribute__((address_space(3))) v4i*)(uintptr_t)(p))

// Copy 8 bf16 (16 bytes) global -> LDS.  Async (GLOBAL_LOAD_ASYNC_TO_LDS_B128,
// per-lane addresses, tracked by ASYNCcnt) when available, else scalar.
__device__ __forceinline__ void copy8_bf16(const __bf16* __restrict__ g,
                                           __bf16* __restrict__ l) {
#if defined(USE_ASYNC_LDS)
  __builtin_amdgcn_global_load_async_to_lds_b128(AS1P(g), AS3P(l), 0, 0);
#else
#pragma unroll
  for (int i = 0; i < 8; ++i) l[i] = g[i];
#endif
}

__device__ __forceinline__ void async_wait_all() {
#if defined(USE_ASYNC_LDS)
#  if __has_builtin(__builtin_amdgcn_s_wait_asynccnt)
  __builtin_amdgcn_s_wait_asynccnt(0);
#  else
  asm volatile("s_wait_asynccnt 0" ::: "memory");
#  endif
#endif
}

// ---- WMMA fragment helpers (CDNA5 ISA 7.12.2 layouts, wave32) -------------
// 16-bit A matrix 16x32: lane l in [0,15] -> row M=l, holds K={0..7,16..23};
// lane l in [16,31] -> row M=l-16, holds K={8..15,24..31}.
__device__ __forceinline__ int akof(int i, int hi) {
  int g = i >> 1, s = i & 1;
  int k = (g < 4) ? (g * 2 + s) : (16 + (g - 4) * 2 + s);
  return k + hi * 8;
}

// A-fragment (16 rows x 32 k) from a row-major LDS tile [row][ld].
// Also used for B tiles stored [n][k] (symmetric layout, N across lanes).
__device__ __forceinline__ v16bf frag_rowmajor(const __bf16* base, int ld, int lane) {
  const int m = lane & 15, hi = (lane >> 4) & 1;
  const __bf16* row = base + m * ld;
  v16bf v;
#pragma unroll
  for (int i = 0; i < 16; ++i) v[i] = row[akof(i, hi)];
  return v;
}

// B-fragment (32 k x 16 n) from an LDS tile stored k-major [k][ld].
__device__ __forceinline__ v16bf frag_kmajor(const __bf16* base, int ld, int lane) {
  const int n = lane & 15, hi = (lane >> 4) & 1;
  v16bf v;
#pragma unroll
  for (int i = 0; i < 16; ++i) v[i] = base[akof(i, hi) * ld + n];
  return v;
}

__device__ __forceinline__ v8f bfmma(v16bf a, v16bf b, v8f c) {
  // v_wmma_f32_16x16x32_bf16: D = A*B + C, f32 accumulate
  return __builtin_amdgcn_wmma_f32_16x16x32_bf16(false, a, false, b,
                                                 (short)0, c, false, false);
}

#define LDA 40  // bf16 LDS pitch for 32-wide K tiles (80 B rows, 16B aligned)
#define LDB 72  // bf16 LDS pitch for 64-wide N tiles (144 B rows, 16B aligned)

// ---------------------------------------------------------------------------
// Kernel 1: Out(bf16) = scale * X(f32, MxK) @ W(f32, KxN).  Block tile 128x64,
// 8 waves (256 thr), wave tile 32x32 (2x2 WMMA).
// ---------------------------------------------------------------------------
__global__ void __launch_bounds__(256)
k_proj(const float* __restrict__ X, const float* __restrict__ W,
       __bf16* __restrict__ Out, int K, int N, float scale)
{
  __shared__ __bf16 As[128 * LDA];
  __shared__ __bf16 Bs[32 * LDB];
  const int tid = threadIdx.x, lane = tid & 31, w = tid >> 5;
  const int wm = w & 3, wn = w >> 2;
  const int m0 = blockIdx.y * 128, n0 = blockIdx.x * 64;
  v8f zero = {};
  v8f c[2][2] = {{zero, zero}, {zero, zero}};

  for (int k0 = 0; k0 < K; k0 += 32) {
    {
      const int row = tid >> 1, col = (tid & 1) * 16;
      const float* src = X + (size_t)(m0 + row) * K + k0 + col;
      __bf16* dst = As + row * LDA + col;
      __builtin_prefetch(src + 32, 0, 0);
#pragma unroll
      for (int i = 0; i < 16; ++i) dst[i] = (__bf16)(src[i] * scale);
    }
    {
      const int row = tid >> 3, col = (tid & 7) * 8;
      const float* src = W + (size_t)(k0 + row) * N + n0 + col;
      __bf16* dst = Bs + row * LDB + col;
#pragma unroll
      for (int i = 0; i < 8; ++i) dst[i] = (__bf16)src[i];
    }
    __syncthreads();
    const __bf16* ab = As + (wm * 32) * LDA;
    const __bf16* bb = Bs + wn * 32;
    v16bf a0 = frag_rowmajor(ab, LDA, lane);
    v16bf a1 = frag_rowmajor(ab + 16 * LDA, LDA, lane);
    v16bf b0 = frag_kmajor(bb, LDB, lane);
    v16bf b1 = frag_kmajor(bb + 16, LDB, lane);
    c[0][0] = bfmma(a0, b0, c[0][0]);
    c[0][1] = bfmma(a0, b1, c[0][1]);
    c[1][0] = bfmma(a1, b0, c[1][0]);
    c[1][1] = bfmma(a1, b1, c[1][1]);
    __syncthreads();
  }
  const int nl = lane & 15, hi = lane >> 4;
#pragma unroll
  for (int tm = 0; tm < 2; ++tm)
#pragma unroll
    for (int tn = 0; tn < 2; ++tn) {
      const int gm = m0 + wm * 32 + tm * 16 + hi * 8;
      const int gn = n0 + wn * 32 + tn * 16 + nl;
#pragma unroll
      for (int r = 0; r < 8; ++r)
        Out[(size_t)(gm + r) * N + gn] = (__bf16)c[tm][tn][r];
    }
}

// ---------------------------------------------------------------------------
// Kernel 2: qEk[b,h,q,r] = sum_d Qs[b,q,h*64+d] * Ek[r,d]  (Q already scaled)
// A = Q slice (bf16, async-copied), B = Ek^T staged [n=r][k=d]. N pad 17*64.
// ---------------------------------------------------------------------------
__global__ void __launch_bounds__(256)
k_qek(const __bf16* __restrict__ Qb, const float* __restrict__ Ek,
      float* __restrict__ qek)
{
  __shared__ __bf16 As[128 * LDA];
  __shared__ __bf16 Bs[64 * LDA];  // [n][k]
  const int tid = threadIdx.x, lane = tid & 31, w = tid >> 5;
  const int wm = w & 3, wn = w >> 2;
  const int z = blockIdx.z, b = z >> 3, h = z & 7;
  const int q0 = blockIdx.y * 128, n0 = blockIdx.x * 64;
  v8f zero = {};
  v8f c[2][2] = {{zero, zero}, {zero, zero}};

  for (int k0 = 0; k0 < HDN; k0 += 32) {
    {
      const int row = tid >> 1, col = (tid & 1) * 16;
      const __bf16* src = Qb + ((size_t)b * SN + q0 + row) * DN + h * HDN + k0 + col;
      __bf16* dst = As + row * LDA + col;
      copy8_bf16(src, dst);
      copy8_bf16(src + 8, dst + 8);
    }
    {
      const int nrow = tid >> 2, col = (tid & 3) * 8;
      const int gr = n0 + nrow;
      __bf16* dst = Bs + nrow * LDA + col;
      if (gr < NRN) {
        const float* src = Ek + (size_t)gr * HDN + k0 + col;
#pragma unroll
        for (int i = 0; i < 8; ++i) dst[i] = (__bf16)src[i];
      } else {
#pragma unroll
        for (int i = 0; i < 8; ++i) dst[i] = (__bf16)0.0f;
      }
    }
    async_wait_all();
    __syncthreads();
    const __bf16* ab = As + (wm * 32) * LDA;
    const __bf16* bb = Bs + (wn * 32) * LDA;
    v16bf a0 = frag_rowmajor(ab, LDA, lane);
    v16bf a1 = frag_rowmajor(ab + 16 * LDA, LDA, lane);
    v16bf b0 = frag_rowmajor(bb, LDA, lane);
    v16bf b1 = frag_rowmajor(bb + 16 * LDA, LDA, lane);
    c[0][0] = bfmma(a0, b0, c[0][0]);
    c[0][1] = bfmma(a0, b1, c[0][1]);
    c[1][0] = bfmma(a1, b0, c[1][0]);
    c[1][1] = bfmma(a1, b1, c[1][1]);
    __syncthreads();
  }
  const int nl = lane & 15, hi = lane >> 4;
#pragma unroll
  for (int tm = 0; tm < 2; ++tm)
#pragma unroll
    for (int tn = 0; tn < 2; ++tn) {
      const int gq = q0 + wm * 32 + tm * 16 + hi * 8;
      const int gn = n0 + wn * 32 + tn * 16 + nl;
      if (gn < NRN) {
#pragma unroll
        for (int r = 0; r < 8; ++r)
          qek[((size_t)z * SN + gq + r) * NRN + gn] = c[tm][tn][r];
      }
    }
}

// ---------------------------------------------------------------------------
// Kernel 3: raw scores alpha[b,h,q,k] = Qs·K + qEk[q, clip(k-q)+512]
// written unnormalized to the attn output region of d_out.
// Q/K tiles are bf16 in global -> async LDS copies.
// ---------------------------------------------------------------------------
__global__ void __launch_bounds__(256)
k_scores(const __bf16* __restrict__ Qb, const __bf16* __restrict__ Kb,
         const float* __restrict__ qek, float* __restrict__ attn)
{
  __shared__ __bf16 As[128 * LDA];  // Q tile [q][d]
  __shared__ __bf16 Bs[64 * LDA];   // K tile [kidx][d]
  const int tid = threadIdx.x, lane = tid & 31, w = tid >> 5;
  const int wm = w & 3, wn = w >> 2;
  const int z = blockIdx.z, b = z >> 3, h = z & 7;
  const int q0 = blockIdx.y * 128, n0 = blockIdx.x * 64;
  v8f zero = {};
  v8f c[2][2] = {{zero, zero}, {zero, zero}};

  for (int k0 = 0; k0 < HDN; k0 += 32) {
    {
      const int row = tid >> 1, col = (tid & 1) * 16;
      const __bf16* src = Qb + ((size_t)b * SN + q0 + row) * DN + h * HDN + k0 + col;
      __bf16* dst = As + row * LDA + col;
      copy8_bf16(src, dst);
      copy8_bf16(src + 8, dst + 8);
    }
    {
      const int nrow = tid >> 2, col = (tid & 3) * 8;
      const __bf16* src = Kb + ((size_t)b * SN + n0 + nrow) * DN + h * HDN + k0 + col;
      __bf16* dst = Bs + nrow * LDA + col;
      copy8_bf16(src, dst);
    }
    async_wait_all();
    __syncthreads();
    const __bf16* ab = As + (wm * 32) * LDA;
    const __bf16* bb = Bs + (wn * 32) * LDA;
    v16bf a0 = frag_rowmajor(ab, LDA, lane);
    v16bf a1 = frag_rowmajor(ab + 16 * LDA, LDA, lane);
    v16bf b0 = frag_rowmajor(bb, LDA, lane);
    v16bf b1 = frag_rowmajor(bb + 16 * LDA, LDA, lane);
    c[0][0] = bfmma(a0, b0, c[0][0]);
    c[0][1] = bfmma(a0, b1, c[0][1]);
    c[1][0] = bfmma(a1, b0, c[1][0]);
    c[1][1] = bfmma(a1, b1, c[1][1]);
    __syncthreads();
  }
  const int nl = lane & 15, hi = lane >> 4;
#pragma unroll
  for (int tm = 0; tm < 2; ++tm)
#pragma unroll
    for (int tn = 0; tn < 2; ++tn) {
      const int gq = q0 + wm * 32 + tm * 16 + hi * 8;
      const int gk = n0 + wn * 32 + tn * 16 + nl;
#pragma unroll
      for (int r = 0; r < 8; ++r) {
        const int q = gq + r;
        int rel = gk - q;
        rel = (rel < -MRN) ? -MRN : (rel > MRN ? MRN : rel);
        const float v = c[tm][tn][r] + qek[((size_t)z * SN + q) * NRN + rel + MRN];
        attn[((size_t)z * SN + q) * SN + gk] = v;
      }
    }
}

// ---------------------------------------------------------------------------
// Kernel 4: fused row softmax (in place in d_out) + wrel bin construction.
// wrel[q,r] (r in 1..1023) = attn[q, q+r-512]; wrel[q,0] / wrel[q,1024] are
// the clipped prefix/suffix sums.  One block per (b,h,q) row.
// ---------------------------------------------------------------------------
__global__ void __launch_bounds__(256)
k_softmax_wrel(float* __restrict__ attn, float* __restrict__ wrel)
{
  __shared__ float red[256];
  __shared__ float rowL[SN];
  const int tid = threadIdx.x;
  const int row = blockIdx.x;          // (b*H+h)*S + q
  const int q = row & (SN - 1);
  float* arow = attn + (size_t)row * SN;

  float vals[8];
  float m = -3.4e38f;
#pragma unroll
  for (int i = 0; i < 8; ++i) { vals[i] = arow[tid + i * 256]; m = fmaxf(m, vals[i]); }
  red[tid] = m; __syncthreads();
  for (int s = 128; s > 0; s >>= 1) { if (tid < s) red[tid] = fmaxf(red[tid], red[tid + s]); __syncthreads(); }
  m = red[0]; __syncthreads();

  float sum = 0.f;
#pragma unroll
  for (int i = 0; i < 8; ++i) { vals[i] = __expf(vals[i] - m); sum += vals[i]; }
  red[tid] = sum; __syncthreads();
  for (int s = 128; s > 0; s >>= 1) { if (tid < s) red[tid] += red[tid + s]; __syncthreads(); }
  const float inv = 1.0f / red[0];
  __syncthreads();

  float lowS = 0.f, highS = 0.f;
#pragma unroll
  for (int i = 0; i < 8; ++i) {
    const int k = tid + i * 256;
    const float v = vals[i] * inv;
    arow[k] = v;
    rowL[k] = v;
    if (k <= q - MRN) lowS += v;   // rel clipped to -512 -> bin 0
    if (k >= q + MRN) highS += v;  // rel clipped to +512 -> bin 1024
  }
  red[tid] = lowS; __syncthreads();
  for (int s = 128; s > 0; s >>= 1) { if (tid < s) red[tid] += red[tid + s]; __syncthreads(); }
  lowS = red[0]; __syncthreads();
  red[tid] = highS; __syncthreads();
  for (int s = 128; s > 0; s >>= 1) { if (tid < s) red[tid] += red[tid + s]; __syncthreads(); }
  highS = red[0]; __syncthreads();

  float* wr = wrel + (size_t)row * NRN;
  for (int r = tid; r < NRN; r += 256) {
    float v;
    if (r == 0)            v = lowS;
    else if (r == NRN - 1) v = highS;
    else { const int j = q + r - MRN; v = (j >= 0 && j < SN) ? rowL[j] : 0.f; }
    wr[r] = v;
  }
}

// ---------------------------------------------------------------------------
// Kernel 5: ctx_h = attn @ V_h  +  wrel @ Ev   (both accumulated via WMMA)
// V tile is bf16 in global -> async LDS copy.
// ---------------------------------------------------------------------------
__global__ void __launch_bounds__(256)
k_pv(const float* __restrict__ attn, const __bf16* __restrict__ Vb,
     const float* __restrict__ wrel, const float* __restrict__ Ev,
     float* __restrict__ ctx)
{
  __shared__ __bf16 As[128 * LDA];
  __shared__ __bf16 Bs[32 * LDB];
  const int tid = threadIdx.x, lane = tid & 31, w = tid >> 5;
  const int wm = w & 3, wn = w >> 2;
  const int z = blockIdx.z, b = z >> 3, h = z & 7;
  const int q0 = blockIdx.y * 128;
  v8f zero = {};
  v8f c[2][2] = {{zero, zero}, {zero, zero}};

  // Phase 1: attn (f32->bf16) @ V_h (bf16), K = S
  for (int k0 = 0; k0 < SN; k0 += 32) {
    {
      const int row = tid >> 1, col = (tid & 1) * 16;
      const float* src = attn + ((size_t)z * SN + q0 + row) * SN + k0 + col;
      __bf16* dst = As + row * LDA + col;
      __builtin_prefetch(src + 32, 0, 0);
#pragma unroll
      for (int i = 0; i < 16; ++i) dst[i] = (__bf16)src[i];
    }
    {
      const int krow = tid >> 3, col = (tid & 7) * 8;
      const __bf16* src = Vb + ((size_t)b * SN + k0 + krow) * DN + h * HDN + col;
      copy8_bf16(src, Bs + krow * LDB + col);
    }
    async_wait_all();
    __syncthreads();
    const __bf16* ab = As + (wm * 32) * LDA;
    const __bf16* bb = Bs + wn * 32;
    v16bf a0 = frag_rowmajor(ab, LDA, lane);
    v16bf a1 = frag_rowmajor(ab + 16 * LDA, LDA, lane);
    v16bf b0 = frag_kmajor(bb, LDB, lane);
    v16bf b1 = frag_kmajor(bb + 16, LDB, lane);
    c[0][0] = bfmma(a0, b0, c[0][0]);
    c[0][1] = bfmma(a0, b1, c[0][1]);
    c[1][0] = bfmma(a1, b0, c[1][0]);
    c[1][1] = bfmma(a1, b1, c[1][1]);
    __syncthreads();
  }

  // Phase 2: wrel (f32->bf16) @ Ev (f32->bf16), K padded 1025 -> 1056
  for (int k0 = 0; k0 < 1056; k0 += 32) {
    {
      const int row = tid >> 1, col = (tid & 1) * 16;
      const float* src = wrel + ((size_t)z * SN + q0 + row) * NRN;
      __bf16* dst = As + row * LDA + col;
#pragma unroll
      for (int i = 0; i < 16; ++i) {
        const int rr = k0 + col + i;
        dst[i] = (rr < NRN) ? (__bf16)src[rr] : (__bf16)0.0f;
      }
    }
    {
      const int krow = tid >> 3, col = (tid & 7) * 8;
      const int rr = k0 + krow;
      __bf16* dst = Bs + krow * LDB + col;
      if (rr < NRN) {
        const float* src = Ev + (size_t)rr * HDN + col;
#pragma unroll
        for (int i = 0; i < 8; ++i) dst[i] = (__bf16)src[i];
      } else {
#pragma unroll
        for (int i = 0; i < 8; ++i) dst[i] = (__bf16)0.0f;
      }
    }
    __syncthreads();
    const __bf16* ab = As + (wm * 32) * LDA;
    const __bf16* bb = Bs + wn * 32;
    v16bf a0 = frag_rowmajor(ab, LDA, lane);
    v16bf a1 = frag_rowmajor(ab + 16 * LDA, LDA, lane);
    v16bf b0 = frag_kmajor(bb, LDB, lane);
    v16bf b1 = frag_kmajor(bb + 16, LDB, lane);
    c[0][0] = bfmma(a0, b0, c[0][0]);
    c[0][1] = bfmma(a0, b1, c[0][1]);
    c[1][0] = bfmma(a1, b0, c[1][0]);
    c[1][1] = bfmma(a1, b1, c[1][1]);
    __syncthreads();
  }

  const int nl = lane & 15, hi = lane >> 4;
#pragma unroll
  for (int tm = 0; tm < 2; ++tm)
#pragma unroll
    for (int tn = 0; tn < 2; ++tn) {
      const int gq = q0 + wm * 32 + tm * 16 + hi * 8;
      const int gn = wn * 32 + tn * 16 + nl;
#pragma unroll
      for (int r = 0; r < 8; ++r)
        ctx[((size_t)b * SN + gq + r) * DN + h * HDN + gn] = c[tm][tn][r];
    }
}

// ---------------------------------------------------------------------------
// Kernel 6: out = ctx @ Wo + bo (f32 out)
// ---------------------------------------------------------------------------
__global__ void __launch_bounds__(256)
k_outproj(const float* __restrict__ X, const float* __restrict__ W,
          const float* __restrict__ bias, float* __restrict__ Out)
{
  __shared__ __bf16 As[128 * LDA];
  __shared__ __bf16 Bs[32 * LDB];
  const int tid = threadIdx.x, lane = tid & 31, w = tid >> 5;
  const int wm = w & 3, wn = w >> 2;
  const int m0 = blockIdx.y * 128, n0 = blockIdx.x * 64;
  v8f zero = {};
  v8f c[2][2] = {{zero, zero}, {zero, zero}};

  for (int k0 = 0; k0 < DN; k0 += 32) {
    {
      const int row = tid >> 1, col = (tid & 1) * 16;
      const float* src = X + (size_t)(m0 + row) * DN + k0 + col;
      __bf16* dst = As + row * LDA + col;
#pragma unroll
      for (int i = 0; i < 16; ++i) dst[i] = (__bf16)src[i];
    }
    {
      const int row = tid >> 3, col = (tid & 7) * 8;
      const float* src = W + (size_t)(k0 + row) * DN + n0 + col;
      __bf16* dst = Bs + row * LDB + col;
#pragma unroll
      for (int i = 0; i < 8; ++i) dst[i] = (__bf16)src[i];
    }
    __syncthreads();
    const __bf16* ab = As + (wm * 32) * LDA;
    const __bf16* bb = Bs + wn * 32;
    v16bf a0 = frag_rowmajor(ab, LDA, lane);
    v16bf a1 = frag_rowmajor(ab + 16 * LDA, LDA, lane);
    v16bf b0 = frag_kmajor(bb, LDB, lane);
    v16bf b1 = frag_kmajor(bb + 16, LDB, lane);
    c[0][0] = bfmma(a0, b0, c[0][0]);
    c[0][1] = bfmma(a0, b1, c[0][1]);
    c[1][0] = bfmma(a1, b0, c[1][0]);
    c[1][1] = bfmma(a1, b1, c[1][1]);
    __syncthreads();
  }
  const int nl = lane & 15, hi = lane >> 4;
#pragma unroll
  for (int tm = 0; tm < 2; ++tm)
#pragma unroll
    for (int tn = 0; tn < 2; ++tn) {
      const int gm = m0 + wm * 32 + tm * 16 + hi * 8;
      const int gn = n0 + wn * 32 + tn * 16 + nl;
      const float bv = bias[gn];
#pragma unroll
      for (int r = 0; r < 8; ++r)
        Out[(size_t)(gm + r) * DN + gn] = c[tm][tn][r] + bv;
    }
}

// ---------------------------------------------------------------------------
extern "C" void kernel_launch(void* const* d_in, const int* in_sizes, int n_in,
                              void* d_out, int out_size, void* d_ws, size_t ws_size,
                              hipStream_t stream)
{
  (void)in_sizes; (void)n_in; (void)out_size; (void)ws_size;
  const float* query = (const float*)d_in[0];
  const float* value = (const float*)d_in[1];
  const float* Wq    = (const float*)d_in[2];
  const float* Wk    = (const float*)d_in[3];
  const float* Wv    = (const float*)d_in[4];
  const float* Wo    = (const float*)d_in[5];
  const float* bo    = (const float*)d_in[6];
  const float* Ek    = (const float*)d_in[7];
  const float* Ev    = (const float*)d_in[8];
  // d_in[9] = heads (fixed 8)

  float* out  = (float*)d_out;                       // (B,S,D)
  float* attn = out + (size_t)BN * SN * DN;          // (B,H,S,S)

  // Workspace layout:
  //   Qb/Kb/Vb : bf16, B*S*D each (Q pre-scaled by 1/sqrt(hd))
  //   ctx      : f32,  B*S*D
  //   qek/wrel : f32,  B*H*S*NUM_RPRS (qEk first, overwritten by wrel)
  const size_t nQKV = (size_t)BN * SN * DN;
  char* ws = (char*)d_ws;
  __bf16* Qb = (__bf16*)ws;
  __bf16* Kb = Qb + nQKV;
  __bf16* Vb = Kb + nQKV;
  float*  ctx = (float*)(ws + 3 * nQKV * sizeof(__bf16));
  float*  qek = ctx + nQKV;

  dim3 blk(256);
  const dim3 gProj(DN / 64, (BN * SN) / 128, 1);
  // NOTE: upstream bug in reference — keys come from `value`.
  k_proj<<<gProj, blk, 0, stream>>>(query, Wq, Qb, DN, DN, 0.125f);
  k_proj<<<gProj, blk, 0, stream>>>(value, Wk, Kb, DN, DN, 1.0f);
  k_proj<<<gProj, blk, 0, stream>>>(value, Wv, Vb, DN, DN, 1.0f);

  k_qek<<<dim3(17, SN / 128, BN * HN), blk, 0, stream>>>(Qb, Ek, qek);
  k_scores<<<dim3(SN / 64, SN / 128, BN * HN), blk, 0, stream>>>(Qb, Kb, qek, attn);
  k_softmax_wrel<<<dim3(BN * HN * SN), blk, 0, stream>>>(attn, qek);
  k_pv<<<dim3(1, SN / 128, BN * HN), blk, 0, stream>>>(attn, Vb, qek, Ev, ctx);
  k_outproj<<<dim3(DN / 64, (BN * SN) / 128, 1), blk, 0, stream>>>(ctx, Wo, bo, out);
}
